// Generator_23545010716903
// MI455X (gfx1250) — matrix operand order repaired and checked by reading, not compiled, not run on hardware
//
#include <hip/hip_runtime.h>
#include <cstdint>
#include <cstddef>

// ---------- problem constants ----------
#define Bq   32
#define Sq   32
#define Eq   512
#define Hq   8
#define DHq  64
#define Vq   10000
#define NBq  4
#define EEq  (Eq*Eq)

typedef __bf16 bf16;
typedef __attribute__((ext_vector_type(16))) __bf16 v16bf;
typedef __attribute__((ext_vector_type(8)))  float  v8f;
typedef __attribute__((ext_vector_type(4)))  unsigned int u32x4;

__device__ __forceinline__ bf16 f2bf(float f) {
    union { float f; uint32_t u; } x; x.f = f;
    uint32_t r = x.u + 0x7FFFu + ((x.u >> 16) & 1u);   // round-to-nearest-even
    union { unsigned short s; bf16 b; } y; y.s = (unsigned short)(r >> 16);
    return y.b;
}

// ---------- positional encoding ----------
__global__ void pe_kernel(float* pe) {
    int idx = blockIdx.x * blockDim.x + threadIdx.x;     // S*E
    if (idx >= Sq * Eq) return;
    int s = idx / Eq, e = idx % Eq;
    int i = e >> 1;
    float dv = __expf(-(float)(2 * i) * (9.210340371976184f / (float)Eq)); // ln(10000)
    float a = (float)s * dv;
    pe[idx] = (e & 1) ? __cosf(a) : __sinf(a);
}

// ---------- tiny (B,S)x(S,S) matmul for the w chain ----------
__global__ void win_kernel(const float* __restrict__ in, const float* __restrict__ W,
                           const float* __restrict__ bias, float* __restrict__ out) {
    int idx = blockIdx.x * blockDim.x + threadIdx.x;     // B*S
    if (idx >= Bq * Sq) return;
    int b = idx / Sq, s = idx % Sq;
    float acc = bias[s];
    for (int k = 0; k < Sq; ++k) acc += in[b * Sq + k] * W[k * Sq + s];
    out[idx] = acc;
}

// ---------- column sums of Wca[n][1] / Wca[n][2] (rank-1 KV cache) ----------
__global__ void colsum_kernel(const float* __restrict__ Wca, float* __restrict__ cs) {
    int idx = blockIdx.x * blockDim.x + threadIdx.x;     // NB*2*E
    if (idx >= NBq * 2 * Eq) return;
    int e = idx % Eq, r = (idx / Eq) & 1, n = idx / (2 * Eq);
    const float* w = Wca + ((size_t)(n * 4 + 1 + r)) * EEq;
    float s = 0.f;
    for (int k = 0; k < Eq; ++k) s += w[(size_t)k * Eq + e];
    cs[idx] = s;
}

__global__ void kvcache_kernel(const float* __restrict__ w, const float* __restrict__ cs,
                               const float* __restrict__ bca, float* __restrict__ Kc,
                               float* __restrict__ Vc) {
    long long idx = (long long)blockIdx.x * blockDim.x + threadIdx.x;  // 2*NB*B*S*E
    if (idx >= (long long)2 * NBq * Bq * Sq * Eq) return;
    int e = idx & (Eq - 1); long long t = idx >> 9;
    int s = t & 31; t >>= 5;
    int b = t & 31; t >>= 5;
    int n = t & 3;  int which = (int)(t >> 2);
    float val = w[b * Sq + s] * cs[(n * 2 + which) * Eq + e] + bca[(n * 4 + 1 + which) * Eq + e];
    float* dst = which ? Vc : Kc;
    dst[(((size_t)n * Bq + b) * Sq + s) * Eq + e] = val;
}

// ---------- weight swizzle into WMMA B-fragment layout ----------
// dst[((kt*Ntiles+nt)*32+lane)*16+i] = W[kt*32 + (lane>>4)*16 + i][nt*16 + (lane&15)]
__global__ void swizzle_b(const float* __restrict__ W, bf16* __restrict__ dst, int K, int N) {
    int idx = blockIdx.x * blockDim.x + threadIdx.x;
    if (idx >= K * N) return;
    int i = idx & 15;
    int lane = (idx >> 4) & 31;
    int tile = idx >> 9;
    int ntiles = N >> 4;
    int nt = tile % ntiles, kt = tile / ntiles;
    int k = kt * 32 + ((lane >> 4) << 4) + i;
    int n = nt * 16 + (lane & 15);
    dst[idx] = f2bf(W[(size_t)k * N + n]);
}

// ---------- Y init + one-hot step 0 ----------
__global__ void init_y(const float* __restrict__ emb, const float* __restrict__ pe,
                       float* __restrict__ Yf, bf16* __restrict__ Yb) {
    int idx = blockIdx.x * blockDim.x + threadIdx.x;     // B*E
    if (idx >= Bq * Eq) return;
    int b = idx / Eq, e = idx % Eq;
    float v = emb[e] + pe[e];                            // START token = 0, position 0
    Yf[((size_t)b * Sq) * Eq + e] = v;
    Yb[((size_t)b * Sq) * Eq + e] = f2bf(v);
}

__global__ void onehot_kernel(float* __restrict__ out) {
    int idx = blockIdx.x * blockDim.x + threadIdx.x;     // B*V
    if (idx >= Bq * Vq) return;
    int b = idx / Vq, v = idx % Vq;
    out[((size_t)b * Sq) * Vq + v] = (v == 0) ? 1.0f : 0.0f;
}

// ---------- WMMA GEMM: each wave computes a 16 x (GROUP*16) strip ----------
// A fragment is loaded once per K-step and reused across GROUP WMMAs.
template <int GROUP>
__global__ void wmma_gemm(const bf16* __restrict__ A, const bf16* __restrict__ Wp,
                          const float* __restrict__ bias, float* __restrict__ Cf,
                          bf16* __restrict__ Cb, int Mtiles, int Ngroups, int K, int N,
                          int relu) {
    int wave = threadIdx.x >> 5;
    int lane = threadIdx.x & 31;
    int tile = blockIdx.x * (blockDim.x >> 5) + wave;
    if (tile >= Mtiles * Ngroups) return;
    int tm = tile / Ngroups, tg = tile % Ngroups;
    int half = lane >> 4, l15 = lane & 15;
    int Ntiles = N >> 4;

    const bf16* Ap = A + (size_t)(tm * 16 + l15) * K + half * 16;
    const bf16* Bp = Wp + ((size_t)tg * GROUP * 32 + lane) * 16;
    const size_t bstride = (size_t)Ntiles * 512;         // elements per K-tile

    const v8f zero = {0.f, 0.f, 0.f, 0.f, 0.f, 0.f, 0.f, 0.f};
    v8f acc[GROUP];
#pragma unroll
    for (int g = 0; g < GROUP; ++g) acc[g] = zero;

    union Frag { v16bf v; u32x4 q[2]; };
    for (int kt = 0; kt < (K >> 5); ++kt) {
        Frag a;
        a.q[0] = *(const u32x4*)(Ap);
        a.q[1] = *(const u32x4*)(Ap + 8);
        Ap += 32;
        // unconditional speculative prefetch of next K-tile's B fragments
        __builtin_prefetch(Bp + bstride, 0, 1);
#pragma unroll
        for (int g = 0; g < GROUP; ++g) {
            Frag b;
            const bf16* bp = Bp + (size_t)g * (32 * 16);
            b.q[0] = *(const u32x4*)(bp);
            b.q[1] = *(const u32x4*)(bp + 8);
            acc[g] = __builtin_amdgcn_wmma_f32_16x16x32_bf16(false, a.v, false, b.v,
                                                             (short)0, acc[g], false, false);
        }
        Bp += bstride;
    }

#pragma unroll
    for (int g = 0; g < GROUP; ++g) {
        int col = (tg * GROUP + g) * 16 + l15;
        float bv = bias ? bias[col] : 0.f;
#pragma unroll
        for (int r = 0; r < 8; ++r) {
            int row = tm * 16 + half * 8 + r;
            float val = acc[g][r] + bv;
            if (relu) val = fmaxf(val, 0.f);
            if (Cf) Cf[(size_t)row * N + col] = val;
            if (Cb) Cb[(size_t)row * N + col] = f2bf(val);
        }
    }
}

// ---------- attention: one wave per (b,h), lane = query row ----------
__global__ void attn_kernel(const float* __restrict__ Q, const float* __restrict__ Kv,
                            const float* __restrict__ Vv, bf16* __restrict__ O,
                            int T, float scale) {
    int bh = blockIdx.x;
    int b = bh >> 3, h = bh & (Hq - 1);
    int t = threadIdx.x;
    if (t >= T) return;
    const float* q = Q + ((size_t)b * Sq + t) * Eq + h * DHq;
    float sc[Sq];
    float mx = -1e30f;
    for (int s = 0; s < T; ++s) {
        const float* k = Kv + ((size_t)b * Sq + s) * Eq + h * DHq;
        float d = 0.f;
        for (int e = 0; e < DHq; ++e) d += q[e] * k[e];
        d *= scale;
        sc[s] = d;
        mx = fmaxf(mx, d);
    }
    float sum = 0.f;
    for (int s = 0; s < T; ++s) { sc[s] = __expf(sc[s] - mx); sum += sc[s]; }
    float inv = 1.f / sum;
    bf16* o = O + ((size_t)b * Sq + t) * Eq + h * DHq;
    for (int e = 0; e < DHq; ++e) {
        float acc = 0.f;
        for (int s = 0; s < T; ++s) acc += sc[s] * Vv[((size_t)b * Sq + s) * Eq + h * DHq + e];
        o[e] = f2bf(acc * inv);
    }
}

// ---------- residual + LayerNorm (one wave per row) ----------
__global__ void resid_ln(const float* __restrict__ X, const float* __restrict__ Yin,
                         const float* __restrict__ g, const float* __restrict__ bt,
                         float* __restrict__ outf, bf16* __restrict__ outb) {
    int row = blockIdx.x;          // B*S rows
    int lane = threadIdx.x;        // 32 lanes
    float v[16];
    float s = 0.f;
    for (int i = 0; i < 16; ++i) {
        int e = lane + i * 32;
        v[i] = X[(size_t)row * Eq + e] + Yin[(size_t)row * Eq + e];
        s += v[i];
    }
    for (int o = 16; o; o >>= 1) s += __shfl_xor(s, o, 32);
    float mu = s * (1.f / Eq);
    float var = 0.f;
    for (int i = 0; i < 16; ++i) { float d = v[i] - mu; var += d * d; }
    for (int o = 16; o; o >>= 1) var += __shfl_xor(var, o, 32);
    float rs = rsqrtf(var * (1.f / Eq) + 1e-5f);
    for (int i = 0; i < 16; ++i) {
        int e = lane + i * 32;
        float val = g[e] * (v[i] - mu) * rs + bt[e];
        outf[(size_t)row * Eq + e] = val;
        outb[(size_t)row * Eq + e] = f2bf(val);
    }
}

// ---------- extract row (tok-1) -> bf16 ----------
__global__ void row_extract(const float* __restrict__ X, bf16* __restrict__ out, int T) {
    int idx = blockIdx.x * blockDim.x + threadIdx.x;     // B*E
    if (idx >= Bq * Eq) return;
    int b = idx / Eq, e = idx % Eq;
    out[idx] = f2bf(X[((size_t)b * Sq + (T - 1)) * Eq + e]);
}

// ---------- softmax over V + argmax ----------
__global__ void softmax_argmax(const float* __restrict__ logits, float* __restrict__ out,
                               int* __restrict__ ids, int tok) {
    int b = blockIdx.x, tid = threadIdx.x;
    __shared__ float sv[256];
    __shared__ int   si[256];
    float mx = -1e30f; int mi = 0;
    for (int v = tid; v < Vq; v += 256) {
        float l = logits[(size_t)b * Vq + v];
        if (l > mx) { mx = l; mi = v; }
    }
    sv[tid] = mx; si[tid] = mi;
    __syncthreads();
    for (int o = 128; o; o >>= 1) {
        if (tid < o) {
            if (sv[tid + o] > sv[tid] || (sv[tid + o] == sv[tid] && si[tid + o] < si[tid])) {
                sv[tid] = sv[tid + o]; si[tid] = si[tid + o];
            }
        }
        __syncthreads();
    }
    float m = sv[0];
    if (tid == 0) ids[b] = si[0];
    __syncthreads();
    float s = 0.f;
    for (int v = tid; v < Vq; v += 256) s += __expf(logits[(size_t)b * Vq + v] - m);
    sv[tid] = s;
    __syncthreads();
    for (int o = 128; o; o >>= 1) { if (tid < o) sv[tid] += sv[tid + o]; __syncthreads(); }
    float inv = 1.f / sv[0];
    for (int v = tid; v < Vq; v += 256)
        out[((size_t)b * Sq + tok) * Vq + v] = __expf(logits[(size_t)b * Vq + v] - m) * inv;
}

// ---------- append next embedding ----------
__global__ void embed_append(const float* __restrict__ emb, const float* __restrict__ pe,
                             const int* __restrict__ ids, float* __restrict__ Yf,
                             bf16* __restrict__ Yb, int tok) {
    int idx = blockIdx.x * blockDim.x + threadIdx.x;     // B*E
    if (idx >= Bq * Eq) return;
    int b = idx / Eq, e = idx % Eq;
    float v = emb[(size_t)ids[b] * Eq + e] + pe[(size_t)tok * Eq + e];
    Yf[((size_t)b * Sq + tok) * Eq + e] = v;
    Yb[((size_t)b * Sq + tok) * Eq + e] = f2bf(v);
}

// =====================================================================
extern "C" void kernel_launch(void* const* d_in, const int* in_sizes, int n_in,
                              void* d_out, int out_size, void* d_ws, size_t ws_size,
                              hipStream_t stream) {
    const float* noise = (const float*)d_in[0];
    const float* W_in  = (const float*)d_in[1];
    const float* b_in  = (const float*)d_in[2];
    const float* emb   = (const float*)d_in[3];
    const float* Wsa   = (const float*)d_in[4];
    const float* bsa   = (const float*)d_in[5];
    const float* Wca   = (const float*)d_in[6];
    const float* bca   = (const float*)d_in[7];
    const float* Wff   = (const float*)d_in[8];
    const float* bff   = (const float*)d_in[9];
    const float* lng   = (const float*)d_in[10];
    const float* lnb   = (const float*)d_in[11];
    const float* softW = (const float*)d_in[12];
    const float* softb = (const float*)d_in[13];
    float* out = (float*)d_out;

    // ---- workspace carve ----
    char* base = (char*)d_ws;
    size_t off = 0;
    auto alloc = [&](size_t bytes) -> void* {
        off = (off + 255) & ~(size_t)255;
        void* p = base + off;
        off += bytes;
        return p;
    };
    const size_t ACT = (size_t)Bq * Sq * Eq;             // 524288 elements
    float* w0   = (float*)alloc(Bq * Sq * 4);
    float* w1   = (float*)alloc(Bq * Sq * 4);
    float* pe   = (float*)alloc(Sq * Eq * 4);
    float* Yf   = (float*)alloc(ACT * 4);
    bf16*  Yb   = (bf16*) alloc(ACT * 2);
    float* xAf  = (float*)alloc(ACT * 4);
    float* xBf  = (float*)alloc(ACT * 4);
    float* h1f  = (float*)alloc(ACT * 4);
    float* h2f  = (float*)alloc(ACT * 4);
    float* qf   = (float*)alloc(ACT * 4);
    float* kf   = (float*)alloc(ACT * 4);
    float* vf   = (float*)alloc(ACT * 4);
    float* of   = (float*)alloc(ACT * 4);
    bf16*  xAb  = (bf16*) alloc(ACT * 2);
    bf16*  xBb  = (bf16*) alloc(ACT * 2);
    bf16*  h1b  = (bf16*) alloc(ACT * 2);
    bf16*  h2b  = (bf16*) alloc(ACT * 2);
    bf16*  aob  = (bf16*) alloc(ACT * 2);
    bf16*  ffb  = (bf16*) alloc(ACT * 2);
    float* Kc   = (float*)alloc((size_t)NBq * ACT * 4);
    float* Vc   = (float*)alloc((size_t)NBq * ACT * 4);
    float* cs   = (float*)alloc(NBq * 2 * Eq * 4);
    bf16*  swz  = (bf16*) alloc((size_t)32 * EEq * 2);
    bf16*  ssoft= (bf16*) alloc((size_t)Eq * Vq * 2);
    float* logits = (float*)alloc((size_t)Bq * Vq * 4);
    bf16*  tokb = (bf16*) alloc((size_t)Bq * Eq * 2);
    int*   ids  = (int*)  alloc(Bq * 4);

    // ---- one-time setup (recomputed every call; deterministic) ----
    pe_kernel<<<(Sq * Eq + 255) / 256, 256, 0, stream>>>(pe);
    win_kernel<<<(Bq * Sq + 255) / 256, 256, 0, stream>>>(noise, W_in, b_in, w0);
    win_kernel<<<(Bq * Sq + 255) / 256, 256, 0, stream>>>(w0, W_in + Sq * Sq, b_in + Sq, w1);
    colsum_kernel<<<(NBq * 2 * Eq + 255) / 256, 256, 0, stream>>>(Wca, cs);
    {
        long long n = (long long)2 * NBq * Bq * Sq * Eq;
        kvcache_kernel<<<(unsigned)((n + 255) / 256), 256, 0, stream>>>(w1, cs, bca, Kc, Vc);
    }
    // swizzle weights: per block slots 0..3 = SA q,k,v,o; 4 = CA q; 5 = CA o; 6,7 = FF
    const float* wsrc[32];
    for (int n = 0; n < NBq; ++n) {
        for (int j = 0; j < 4; ++j) wsrc[n * 8 + j] = Wsa + (size_t)(n * 4 + j) * EEq;
        wsrc[n * 8 + 4] = Wca + (size_t)(n * 4 + 0) * EEq;
        wsrc[n * 8 + 5] = Wca + (size_t)(n * 4 + 3) * EEq;
        wsrc[n * 8 + 6] = Wff + (size_t)(n * 2 + 0) * EEq;
        wsrc[n * 8 + 7] = Wff + (size_t)(n * 2 + 1) * EEq;
    }
    for (int m = 0; m < 32; ++m)
        swizzle_b<<<EEq / 256, 256, 0, stream>>>(wsrc[m], swz + (size_t)m * EEq, Eq, Eq);
    swizzle_b<<<(Eq * Vq) / 256, 256, 0, stream>>>(softW, ssoft, Eq, Vq);
    init_y<<<(Bq * Eq + 255) / 256, 256, 0, stream>>>(emb, pe, Yf, Yb);
    onehot_kernel<<<(Bq * Vq + 255) / 256, 256, 0, stream>>>(out);

    auto swp = [&](int n, int slot) { return swz + (size_t)(n * 8 + slot) * EEq; };
    auto gemm = [&](const bf16* A, const bf16* W, const float* bias, float* Cf, bf16* Cb,
                    int relu) {
        // Mtiles=64 (1024 rows), Ngroups=8 (each wave does 16x64) -> 512 waves / 8 per block
        wmma_gemm<4><<<64, 256, 0, stream>>>(A, W, bias, Cf, Cb, 64, 8, Eq, Eq, relu);
    };

    const float scale = 0.125f;  // 1/sqrt(64)

    // ---- sequential generation ----
    for (int tok = 1; tok < Sq; ++tok) {
        const float* inF = Yf;
        const bf16*  inB = Yb;
        for (int n = 0; n < NBq; ++n) {
            // self-attention
            gemm(inB, swp(n, 0), bsa + (n * 4 + 0) * Eq, qf, nullptr, 0);
            gemm(inB, swp(n, 1), bsa + (n * 4 + 1) * Eq, kf, nullptr, 0);
            gemm(inB, swp(n, 2), bsa + (n * 4 + 2) * Eq, vf, nullptr, 0);
            attn_kernel<<<Bq * Hq, 32, 0, stream>>>(qf, kf, vf, aob, tok, scale);
            gemm(aob, swp(n, 3), bsa + (n * 4 + 3) * Eq, of, nullptr, 0);
            resid_ln<<<Bq * Sq, 32, 0, stream>>>(inF, of, lng + (n * 3 + 0) * Eq,
                                                 lnb + (n * 3 + 0) * Eq, h1f, h1b);
            // cross-attention (rank-1 KV cache)
            gemm(h1b, swp(n, 4), bca + (n * 4 + 0) * Eq, qf, nullptr, 0);
            attn_kernel<<<Bq * Hq, 32, 0, stream>>>(qf, Kc + (size_t)n * ACT,
                                                    Vc + (size_t)n * ACT, aob, tok, scale);
            gemm(aob, swp(n, 5), bca + (n * 4 + 3) * Eq, of, nullptr, 0);
            resid_ln<<<Bq * Sq, 32, 0, stream>>>(h1f, of, lng + (n * 3 + 1) * Eq,
                                                 lnb + (n * 3 + 1) * Eq, h2f, h2b);
            // feed-forward
            gemm(h2b, swp(n, 6), bff + (n * 2 + 0) * Eq, nullptr, ffb, 1);
            gemm(ffb, swp(n, 7), bff + (n * 2 + 1) * Eq, of, nullptr, 0);
            float* oF = (n & 1) ? xBf : xAf;
            bf16*  oB = (n & 1) ? xBb : xAb;
            resid_ln<<<Bq * Sq, 32, 0, stream>>>(h2f, of, lng + (n * 3 + 2) * Eq,
                                                 lnb + (n * 3 + 2) * Eq, oF, oB);
            inF = oF; inB = oB;
        }
        // vocab projection + softmax + argmax + append
        row_extract<<<(Bq * Eq + 255) / 256, 256, 0, stream>>>(inF, tokb, tok);
        // 2 Mtiles x 125 groups of 5 N-tiles (625*16 = 10000)
        wmma_gemm<5><<<32, 256, 0, stream>>>(tokb, ssoft, softb, logits, nullptr,
                                             2, 125, Eq, Vq, 0);
        softmax_argmax<<<Bq, 256, 0, stream>>>(logits, out, ids, tok);
        embed_append<<<(Bq * Eq + 255) / 256, 256, 0, stream>>>(emb, pe, ids, Yf, Yb, tok);
    }
    (void)in_sizes; (void)n_in; (void)out_size; (void)ws_size;
}
